// SAGELayer_30245159698932
// MI455X (gfx1250) — matrix operand order repaired and checked by reading, not compiled, not run on hardware
//
#include <hip/hip_runtime.h>
#include <hip/hip_bf16.h>

#define D 32
#define DOUT 128
#define NTILES_B 50   // 2 (W111) + 16 (Wmsg) + 32 (Wapp) WMMA B-tiles

typedef __attribute__((ext_vector_type(16))) __bf16        v16bf;
typedef __attribute__((ext_vector_type(8)))  float         v8f;
typedef __attribute__((ext_vector_type(8)))  unsigned int  v8u;

// ---- helpers ---------------------------------------------------------------

// pack two fp32 -> one dword of 2 x bf16 (a -> low half, b -> high half), RNE
__device__ __forceinline__ unsigned int pack_bf2(float a, float b) {
  unsigned int ua = __builtin_bit_cast(unsigned int, a);
  unsigned int ub = __builtin_bit_cast(unsigned int, b);
  ua += 0x7FFFu + ((ua >> 16) & 1u);
  ub += 0x7FFFu + ((ub >> 16) & 1u);
  return (ua >> 16) | (ub & 0xFFFF0000u);
}

// Load a 16x32-bf16 A tile row (one lane's 8 dwords) from a contiguous fp32
// row.  A-layout (ISA 7.12.2): slots 0..7 hold K = group*8+s, slots 8..15 hold
// K = 16+group*8+(s-8) -> two contiguous 8-float runs.
__device__ __forceinline__ v16bf load_a_f32(const float* __restrict__ row,
                                            int group, float scale) {
  const float4* r0 = (const float4*)(row + group * 8);
  const float4* r1 = (const float4*)(row + 16 + group * 8);
  float4 x0 = r0[0], x1 = r0[1], x2 = r1[0], x3 = r1[1];
  v8u raw;
  raw[0] = pack_bf2(scale * x0.x, scale * x0.y);
  raw[1] = pack_bf2(scale * x0.z, scale * x0.w);
  raw[2] = pack_bf2(scale * x1.x, scale * x1.y);
  raw[3] = pack_bf2(scale * x1.z, scale * x1.w);
  raw[4] = pack_bf2(scale * x2.x, scale * x2.y);
  raw[5] = pack_bf2(scale * x2.z, scale * x2.w);
  raw[6] = pack_bf2(scale * x3.x, scale * x3.y);
  raw[7] = pack_bf2(scale * x3.z, scale * x3.w);
  return __builtin_bit_cast(v16bf, raw);
}

// Load a pre-packed WMMA B tile (256 dwords per tile, 8 dwords per lane).
__device__ __forceinline__ v16bf load_btile(const unsigned int* __restrict__ Bp,
                                            int tile, int lane) {
  const v8u* p = (const v8u*)(Bp + tile * 256);
  return __builtin_bit_cast(v16bf, p[lane]);
}

__device__ __forceinline__ v8f wmma_bf16(v16bf a, v16bf b, v8f c) {
  return __builtin_amdgcn_wmma_f32_16x16x32_bf16(
      false, a, false, b, (short)0, c, false, false);
}

// ---- kernel 0: pre-pack all weight matrices into WMMA B-layout bf16 --------
// B[k][n] = W[nbase+n][kbase+k]; lanes 0-15: n=lane, K=0..7,16..23; lanes 16-31:
// n=lane-16, K=8..15,24..31.  One wave per 32x16 tile.

__global__ void k_packw(const float* __restrict__ W111,
                        const float* __restrict__ Wmsg,
                        const float* __restrict__ Wapp,
                        unsigned int* __restrict__ Bp) {
  int wave = (blockIdx.x * blockDim.x + threadIdx.x) >> 5;
  int lane = threadIdx.x & 31;
  if (wave >= NTILES_B) return;
  const float* W; int ldw, nbase, kbase;
  if (wave < 2)       { W = W111; ldw = D;     nbase = wave * 16;        kbase = 0; }
  else if (wave < 18) { int t = wave - 2;  W = Wmsg; ldw = 2 * D;
                        nbase = (t >> 1) * 16; kbase = (t & 1) * 32; }
  else                { int t = wave - 18; W = Wapp; ldw = DOUT;
                        nbase = (t >> 2) * 16; kbase = (t & 3) * 32; }
  int n = lane & 15, kofs = (lane >> 4) * 8;
  const float4* r = (const float4*)(W + (unsigned)(nbase + n) * ldw + kbase + kofs);
  float4 x0 = r[0], x1 = r[1], x2 = r[4], x3 = r[5];
  v8u raw;
  raw[0] = pack_bf2(x0.x, x0.y); raw[1] = pack_bf2(x0.z, x0.w);
  raw[2] = pack_bf2(x1.x, x1.y); raw[3] = pack_bf2(x1.z, x1.w);
  raw[4] = pack_bf2(x2.x, x2.y); raw[5] = pack_bf2(x2.z, x2.w);
  raw[6] = pack_bf2(x3.x, x3.y); raw[7] = pack_bf2(x3.z, x3.w);
  ((v8u*)(Bp + wave * 256))[lane] = raw;
}

// ---- kernel 1: in-degree counts -------------------------------------------

__global__ void k_count(const int* __restrict__ dst, float* __restrict__ cnt,
                        int E) {
  int e = blockIdx.x * blockDim.x + threadIdx.x;
  if (e < E) atomicAdd(cnt + (unsigned)dst[e], 1.0f);
}

// ---- kernel 2: pass1, m1 = (2e) @ W111^T, atomic scatter-add to dst --------

__global__ void k_pass1(const float* __restrict__ efeats,
                        const int* __restrict__ dst,
                        const unsigned int* __restrict__ Bp,
                        float* __restrict__ segsum1, int E) {
  int wave = (blockIdx.x * blockDim.x + threadIdx.x) >> 5;
  int lane = threadIdx.x & 31;
  int base = wave * 16;                 // wave-uniform
  if (base >= E) return;
  bool full = (base + 16 <= E);         // wave-uniform

  __builtin_prefetch(efeats + (unsigned)(base + 16) * D, 0, 0);

  int group = lane >> 4;
  int m     = lane & 15;
  int edge  = base + m;

  v16bf a;
  if (full) {
    a = load_a_f32(efeats + (unsigned)edge * D, group, 2.0f);
  } else {
    unsigned ec = edge < E ? (unsigned)edge : 0u;
    a = load_a_f32(efeats + ec * D, group, edge < E ? 2.0f : 0.0f);
  }

  v16bf b0 = load_btile(Bp, 0, lane);
  v16bf b1 = load_btile(Bp, 1, lane);
  v8f z = {};
  v8f c0 = wmma_bf16(a, b0, z);         // cols 0..15
  v8f c1 = wmma_bf16(a, b1, z);         // cols 16..31

  unsigned col = lane & 15;
  if (full) {
    const int4* dp = (const int4*)(dst + base + 8 * group);
    int4 da = dp[0], db = dp[1];
    int d8[8] = { da.x, da.y, da.z, da.w, db.x, db.y, db.z, db.w };
    unsigned rowofs[8];
#pragma unroll
    for (int r = 0; r < 8; ++r) rowofs[r] = (unsigned)d8[r] * D + col;
#pragma unroll
    for (int r = 0; r < 8; ++r) {
      atomicAdd(segsum1 + rowofs[r],      c0[r]);
      atomicAdd(segsum1 + rowofs[r] + 16, c1[r]);
    }
  } else {
#pragma unroll
    for (int r = 0; r < 8; ++r) {
      int e2 = base + r + 8 * group;
      if (e2 < E) {
        unsigned ofs = (unsigned)dst[e2] * D + col;
        atomicAdd(segsum1 + ofs,      c0[r]);
        atomicAdd(segsum1 + ofs + 16, c1[r]);
      }
    }
  }
}

// ---- kernel 3: h = relu(sum/cnt + b111) if cnt>0 else 0, stored bf16 -------
// One thread per dword (2 features).

__global__ void k_hfin(const float* __restrict__ segsum1,
                       const float* __restrict__ cnt,
                       const float* __restrict__ b111,
                       unsigned int* __restrict__ hbf, int N) {
  int i = blockIdx.x * blockDim.x + threadIdx.x;   // over N*16 dwords
  if (i >= N * (D / 2)) return;
  unsigned n = (unsigned)i >> 4, jp = (unsigned)i & 15;
  float c = cnt[n];
  float v0 = 0.f, v1 = 0.f;
  if (c > 0.f) {
    float inv = 1.0f / c;
    v0 = segsum1[n * D + 2 * jp]     * inv + b111[2 * jp];
    v1 = segsum1[n * D + 2 * jp + 1] * inv + b111[2 * jp + 1];
    v0 = v0 > 0.f ? v0 : 0.f;
    v1 = v1 > 0.f ? v1 : 0.f;
  }
  hbf[i] = pack_bf2(v0, v1);
}

// ---- kernel 4: pass2, m2 = cat(h[src], e) @ Wmsg^T, scatter-add ------------

__global__ void k_pass2(const float* __restrict__ efeats,
                        const unsigned int* __restrict__ hbf, // bf16 pairs
                        const int* __restrict__ src,
                        const int* __restrict__ dst,
                        const unsigned int* __restrict__ Bp,
                        float* __restrict__ segsum2, int E) {
  int wave = (blockIdx.x * blockDim.x + threadIdx.x) >> 5;
  int lane = threadIdx.x & 31;
  int base = wave * 16;                 // wave-uniform
  if (base >= E) return;
  bool full = (base + 16 <= E);

  __builtin_prefetch(efeats + (unsigned)(base + 16) * D, 0, 0);

  int group = lane >> 4;
  int m     = lane & 15;
  int edge  = base + m;
  unsigned ec = edge < E ? (unsigned)edge : 0u;
  float esc = (full || edge < E) ? 1.0f : 0.0f;

  // a1: h[src] gather.  hbf rows are already bf16 in natural K order, so the
  // two 8-element K runs are two 16-byte vector loads.
  unsigned sn = (unsigned)src[ec];
  const uint4* hr0 = (const uint4*)(hbf + sn * (D / 2) + group * 4);
  const uint4* hr1 = (const uint4*)(hbf + sn * (D / 2) + 8 + group * 4);
  uint4 h0 = hr0[0], h1 = hr1[0];
  v8u rawh;
  rawh[0] = h0.x; rawh[1] = h0.y; rawh[2] = h0.z; rawh[3] = h0.w;
  rawh[4] = h1.x; rawh[5] = h1.y; rawh[6] = h1.z; rawh[7] = h1.w;
  if (esc == 0.0f) {
#pragma unroll
    for (int s = 0; s < 8; ++s) rawh[s] = 0u;
  }
  v16bf a1 = __builtin_bit_cast(v16bf, rawh);
  v16bf a2 = load_a_f32(efeats + ec * D, group, esc);

  unsigned col = lane & 15;
  unsigned rowofs[8];
  bool valid[8];
  if (full) {
    const int4* dp = (const int4*)(dst + base + 8 * group);
    int4 da = dp[0], db = dp[1];
    int d8[8] = { da.x, da.y, da.z, da.w, db.x, db.y, db.z, db.w };
#pragma unroll
    for (int r = 0; r < 8; ++r) {
      rowofs[r] = (unsigned)d8[r] * DOUT + col;
      valid[r]  = true;
    }
  } else {
#pragma unroll
    for (int r = 0; r < 8; ++r) {
      int e2 = base + r + 8 * group;
      valid[r]  = (e2 < E);
      rowofs[r] = valid[r] ? ((unsigned)dst[e2] * DOUT + col) : 0u;
    }
  }

#pragma unroll
  for (int nb = 0; nb < 8; ++nb) {      // 8 blocks of 16 output cols
    v16bf bh = load_btile(Bp, 2 + nb * 2,     lane);  // K = 0..31 (h)
    v16bf be = load_btile(Bp, 2 + nb * 2 + 1, lane);  // K = 32..63 (e)
    v8f z = {};
    v8f c = wmma_bf16(a1, bh, z);
    c     = wmma_bf16(a2, be, c);       // chained accumulate over K=64
    if (full) {
#pragma unroll
      for (int r = 0; r < 8; ++r)       // nb*16 folds into the imm offset
        atomicAdd(segsum2 + rowofs[r] + nb * 16, c[r]);
    } else {
#pragma unroll
      for (int r = 0; r < 8; ++r)
        if (valid[r])
          atomicAdd(segsum2 + rowofs[r] + nb * 16, c[r]);
    }
  }
}

// ---- kernel 5: out = relu( (2*h_neigh) @ Wapp^T + bapp ) -------------------

__global__ void k_apply(const float* __restrict__ segsum2,
                        const float* __restrict__ cnt,
                        const float* __restrict__ bmsg,   // [128]
                        const unsigned int* __restrict__ Bp,
                        const float* __restrict__ bapp,   // [128]
                        float* __restrict__ out, int N) {
  int wave = (blockIdx.x * blockDim.x + threadIdx.x) >> 5;
  int lane = threadIdx.x & 31;
  int base = wave * 16;                 // wave-uniform
  if (base >= N) return;
  bool full = (base + 16 <= N);

  int group = lane >> 4;
  int m     = lane & 15;
  int node  = base + m;
  unsigned nc = node < N ? (unsigned)node : 0u;

  float c0   = cnt[nc];
  bool  has  = (node < N) && (c0 > 0.f);
  float inv2 = has ? 2.0f / c0 : 0.0f;  // scale on the sum
  float bs   = has ? 2.0f : 0.0f;       // scale on b_msg

  const float* srow = segsum2 + nc * DOUT;

  // A = 2*(sum/cnt + bmsg) for cnt>0 else 0;  K = 128 in 4 chunks of 32
  v16bf a[4];
#pragma unroll
  for (int kc = 0; kc < 4; ++kc) {
    const float4* s0 = (const float4*)(srow + kc * 32 + group * 8);
    const float4* s1 = (const float4*)(srow + kc * 32 + 16 + group * 8);
    const float4* g0 = (const float4*)(bmsg + kc * 32 + group * 8);
    const float4* g1 = (const float4*)(bmsg + kc * 32 + 16 + group * 8);
    float4 x0 = s0[0], x1 = s0[1], x2 = s1[0], x3 = s1[1];
    float4 y0 = g0[0], y1 = g0[1], y2 = g1[0], y3 = g1[1];
    v8u raw;
    raw[0] = pack_bf2(fmaf(x0.x, inv2, bs * y0.x), fmaf(x0.y, inv2, bs * y0.y));
    raw[1] = pack_bf2(fmaf(x0.z, inv2, bs * y0.z), fmaf(x0.w, inv2, bs * y0.w));
    raw[2] = pack_bf2(fmaf(x1.x, inv2, bs * y1.x), fmaf(x1.y, inv2, bs * y1.y));
    raw[3] = pack_bf2(fmaf(x1.z, inv2, bs * y1.z), fmaf(x1.w, inv2, bs * y1.w));
    raw[4] = pack_bf2(fmaf(x2.x, inv2, bs * y2.x), fmaf(x2.y, inv2, bs * y2.y));
    raw[5] = pack_bf2(fmaf(x2.z, inv2, bs * y2.z), fmaf(x2.w, inv2, bs * y2.w));
    raw[6] = pack_bf2(fmaf(x3.x, inv2, bs * y3.x), fmaf(x3.y, inv2, bs * y3.y));
    raw[7] = pack_bf2(fmaf(x3.z, inv2, bs * y3.z), fmaf(x3.w, inv2, bs * y3.w));
    a[kc] = __builtin_bit_cast(v16bf, raw);
  }

  unsigned col = lane & 15;
  unsigned rowofs[8];
  bool valid[8];
#pragma unroll
  for (int r = 0; r < 8; ++r) {
    int nn = base + r + 8 * group;
    valid[r]  = full || (nn < N);
    rowofs[r] = (valid[r] ? (unsigned)nn : 0u) * DOUT + col;
  }

#pragma unroll
  for (int nb = 0; nb < 8; ++nb) {
    v8f c = {};
#pragma unroll
    for (int kc = 0; kc < 4; ++kc) {
      v16bf b = load_btile(Bp, 18 + nb * 4 + kc, lane);
      c = wmma_bf16(a[kc], b, c);
    }
    float bap = bapp[nb * 16 + col];
    if (full) {
#pragma unroll
      for (int r = 0; r < 8; ++r) {     // nb*16 folds into the imm offset
        float v = c[r] + bap;
        out[rowofs[r] + nb * 16] = v > 0.f ? v : 0.f;
      }
    } else {
#pragma unroll
      for (int r = 0; r < 8; ++r) {
        if (valid[r]) {
          float v = c[r] + bap;
          out[rowofs[r] + nb * 16] = v > 0.f ? v : 0.f;
        }
      }
    }
  }
}

// ---- launcher --------------------------------------------------------------

extern "C" void kernel_launch(void* const* d_in, const int* in_sizes, int n_in,
                              void* d_out, int out_size, void* d_ws, size_t ws_size,
                              hipStream_t stream) {
  // setup_inputs order:
  // 0 nfeats (unused by reference), 1 efeats, 2 W_msg111_w, 3 W_msg111_b,
  // 4 W_msg_w, 5 W_msg_b, 6 W_apply_w, 7 W_apply_b, 8 src, 9 dst
  const float* efeats = (const float*)d_in[1];
  const float* W111   = (const float*)d_in[2];
  const float* b111   = (const float*)d_in[3];
  const float* Wmsg   = (const float*)d_in[4];
  const float* bmsg   = (const float*)d_in[5];
  const float* Wapp   = (const float*)d_in[6];
  const float* bapp   = (const float*)d_in[7];
  const int*   src    = (const int*)d_in[8];
  const int*   dst    = (const int*)d_in[9];

  int E = in_sizes[8];
  int N = in_sizes[0] / D;

  char* ws = (char*)d_ws;
  size_t off = 0;
  unsigned int* Bp = (unsigned int*)(ws + off);
  off += (size_t)NTILES_B * 256 * sizeof(unsigned int);   // 50 KB packed weights
  float* segsum1 = (float*)(ws + off); off += (size_t)N * D * sizeof(float);
  float* cnt     = (float*)(ws + off); off += (size_t)N * sizeof(float);
  off = (off + 255) & ~(size_t)255;
  unsigned int* hbf = (unsigned int*)(ws + off);          // bf16 pairs [N][16]
  off += (size_t)N * (D / 2) * sizeof(unsigned int);
  off = (off + 255) & ~(size_t)255;
  float* segsum2 = (float*)(ws + off); off += (size_t)N * DOUT * sizeof(float);

  hipMemsetAsync(segsum1, 0, (size_t)N * D * sizeof(float), stream);
  hipMemsetAsync(cnt,     0, (size_t)N * sizeof(float), stream);
  hipMemsetAsync(segsum2, 0, (size_t)N * DOUT * sizeof(float), stream);

  int tilesE  = (E + 15) / 16;
  int blocksE = (tilesE + 3) / 4;       // 4 waves (128 threads) per block
  int tilesN  = (N + 15) / 16;
  int blocksN = (tilesN + 3) / 4;

  k_packw<<<(NTILES_B + 3) / 4, 128, 0, stream>>>(W111, Wmsg, Wapp, Bp);
  k_count<<<(E + 255) / 256, 256, 0, stream>>>(dst, cnt, E);
  k_pass1<<<blocksE, 128, 0, stream>>>(efeats, dst, Bp, segsum1, E);
  k_hfin<<<(N * (D / 2) + 255) / 256, 256, 0, stream>>>(segsum1, cnt, b111, hbf, N);
  k_pass2<<<blocksE, 128, 0, stream>>>(efeats, hbf, src, dst, Bp, segsum2, E);
  k_apply<<<blocksN, 128, 0, stream>>>(segsum2, cnt, bmsg, Bp, bapp,
                                       (float*)d_out, N);
}
